// biLSTM_24421184045776
// MI455X (gfx1250) — matrix operand order repaired and checked by reading, not compiled, not run on hardware
//
#include <hip/hip_runtime.h>
#include <hip/hip_bf16.h>
#include <math.h>

// ---------------- problem dims ----------------
constexpr int SEQ = 128;   // S
constexpr int BAT = 256;   // B
constexpr int IN  = 512;   // I
constexpr int HID = 1024;  // H
constexpr int FH  = 4096;  // 4H
constexpr int KK  = 1536;  // I + H
constexpr int NC  = 10;    // classes
constexpr int NKT = KK / 32;   // 48 K-steps

typedef __bf16 bf16_t;
typedef __attribute__((ext_vector_type(16))) __bf16 v16bf;
typedef __attribute__((ext_vector_type(8)))  __bf16 v8bf;
typedef __attribute__((ext_vector_type(8)))  float  v8f;

union AFrag { v16bf v; v8bf h[2]; };

__device__ inline bf16_t f2bf(float f) {
    union { float f; unsigned u; } v; v.f = f;
    unsigned r = v.u + 0x7FFFu + ((v.u >> 16) & 1u);   // round-to-nearest-even
    union { unsigned short s; bf16_t b; } o; o.s = (unsigned short)(r >> 16);
    return o.b;
}

// ---------------- workspace layout (bytes) ----------------
constexpr size_t OFF_W  = 0;                                   // 2*FH*KK bf16   = 25165824
constexpr size_t OFF_X  = OFF_W  + (size_t)2*FH*KK*2;          // SEQ*BAT*IN bf16= 33554432
constexpr size_t OFF_H  = OFF_X  + (size_t)SEQ*BAT*IN*2;       // 2*BAT*HID bf16 =  1048576
constexpr size_t OFF_C  = OFF_H  + (size_t)2*BAT*HID*2;        // 2*BAT*HID f32  =  2097152
constexpr size_t OFF_Z  = OFF_C  + (size_t)2*BAT*HID*4;        // 2*BAT*FH  f32  =  8388608
constexpr size_t OFF_HC = OFF_Z  + (size_t)2*BAT*FH*4;         // BAT*2H    f32  =  2097152

// ---------------- prep kernels (8-wide vectorized) ----------------
__global__ void cvt_weights(const float* __restrict__ fw, const float* __restrict__ bw,
                            bf16_t* __restrict__ Wbf) {
    size_t i8 = (size_t)blockIdx.x * blockDim.x + threadIdx.x;   // over 2*FH*KK/8
    const size_t NW8 = (size_t)FH * KK / 8;
    if (i8 >= 2 * NW8) return;
    const float* src = (i8 < NW8) ? (fw + i8 * 8) : (bw + (i8 - NW8) * 8);
    bf16_t tmp[8];
    #pragma unroll
    for (int e = 0; e < 8; ++e) tmp[e] = f2bf(src[e]);
    *(uint4*)(Wbf + i8 * 8) = *(const uint4*)tmp;
}

__global__ void gather_embed(const int* __restrict__ x, const float* __restrict__ embed,
                             bf16_t* __restrict__ Xbf) {
    size_t i8 = (size_t)blockIdx.x * blockDim.x + threadIdx.x;   // over SEQ*BAT*IN/8
    if (i8 >= (size_t)SEQ * BAT * IN / 8) return;
    int s   = (int)(i8 >> 14);           // BAT*IN/8 = 2^14
    int rem = (int)(i8 & 16383u);
    int b   = rem >> 6;                  // IN/8 = 64
    int k8  = (rem & 63) << 3;
    int tok = x[b * SEQ + s];            // x is [B,S]
    const float* src = embed + (size_t)tok * IN + k8;
    bf16_t tmp[8];
    #pragma unroll
    for (int e = 0; e < 8; ++e) tmp[e] = f2bf(src[e]);
    *(uint4*)(Xbf + i8 * 8) = *(const uint4*)tmp;
}

__global__ void init_state(float* __restrict__ Cst, bf16_t* __restrict__ Hbf) {
    size_t idx = (size_t)blockIdx.x * blockDim.x + threadIdx.x;   // over 2*BAT*HID
    if (idx >= (size_t)2 * BAT * HID) return;
    Cst[idx] = 0.f;
    Hbf[idx] = f2bf(0.f);
}

// ---------------- per-step GEMM: Z = [x_t | h] @ W^T + b (both directions) ----------------
// Block: 128 thr (4 waves). Block tile 64(M) x 128(N), K-step 32, double-buffered LDS
// filled by global_load_async_to_lds_b128 (ASYNCcnt). Wave tile 32x64 -> 8 WMMA / K-step.
constexpr int LDT = 40;   // padded LDS row stride (80B: 16B aligned, bank-spread)

__global__ __launch_bounds__(128)
void lstm_step_gemm(const bf16_t* __restrict__ Xbf,
                    const bf16_t* __restrict__ Hbf,
                    const bf16_t* __restrict__ Wbf,
                    const float* __restrict__ fwd_b,
                    const float* __restrict__ bwd_b,
                    float* __restrict__ Z, int t)
{
    __shared__ __attribute__((aligned(16))) bf16_t lA[2][64 * LDT];
    __shared__ __attribute__((aligned(16))) bf16_t lB[2][128 * LDT];

    const int dir     = blockIdx.z;
    const int colBase = blockIdx.x * 128;         // over FH
    const int rowBase = blockIdx.y * 64;          // over BAT
    const int tid  = threadIdx.x;
    const int lane = tid & 31;
    const int w    = tid >> 5;                    // 4 waves
    const int wRow = (w >> 1) * 32;
    const int wCol = (w & 1) * 64;
    const int g    = lane >> 4;                   // lane group 0/1
    const int rl   = lane & 15;
    const int trow = dir ? (SEQ - 1 - t) : t;

    const bf16_t* Wd   = Wbf + (size_t)dir * FH * KK;
    const bf16_t* Hd   = Hbf + (size_t)dir * BAT * HID;
    const float*  bias = dir ? bwd_b : fwd_b;
    float*        Zd   = Z + (size_t)dir * BAT * FH;

    // ---- per-thread staging geometry (16B chunks; 2 for A, 4 for B) ----
    const int rA = tid >> 2;                      // 0..31
    const int ck = (tid & 3) << 3;                // 0,8,16,24 bf16 elems
    const bf16_t* gA0x = Xbf + ((size_t)trow * BAT + rowBase + rA) * IN + ck;
    const bf16_t* gA1x = gA0x + (size_t)32 * IN;
    const bf16_t* gA0h = Hd + (size_t)(rowBase + rA) * HID + ck;
    const bf16_t* gA1h = gA0h + (size_t)32 * HID;
    const bf16_t* gB0  = Wd + (size_t)(colBase + rA) * KK + ck;   // +32*KK per extra row block
    const unsigned ldsA0 = (unsigned)(size_t)&lA[0][rA * LDT + ck];
    const unsigned ldsB0 = (unsigned)(size_t)&lB[0][rA * LDT + ck];
    constexpr unsigned bufStrideA = 64u * LDT * 2u;
    constexpr unsigned bufStrideB = 128u * LDT * 2u;
    constexpr unsigned rowBlkLds  = 32u * LDT * 2u;

    auto stage = [&](int k0, int buf) {
        const bf16_t* a0;
        const bf16_t* a1;
        if (k0 < IN) { a0 = gA0x + k0;        a1 = gA1x + k0; }
        else         { a0 = gA0h + (k0 - IN); a1 = gA1h + (k0 - IN); }
        const unsigned dA = ldsA0 + (unsigned)buf * bufStrideA;
        asm volatile("global_load_async_to_lds_b128 %0, %1, off"
                     :: "v"(dA),             "v"(a0) : "memory");
        asm volatile("global_load_async_to_lds_b128 %0, %1, off"
                     :: "v"(dA + rowBlkLds), "v"(a1) : "memory");
        const unsigned dB = ldsB0 + (unsigned)buf * bufStrideB;
        const bf16_t* bsrc = gB0 + k0;
        #pragma unroll
        for (int i = 0; i < 4; ++i) {
            asm volatile("global_load_async_to_lds_b128 %0, %1, off"
                         :: "v"(dB + (unsigned)i * rowBlkLds),
                        "v"(bsrc + (size_t)i * 32 * KK) : "memory");
        }
    };

    v8f acc[2][4];
    #pragma unroll
    for (int m = 0; m < 2; ++m)
        #pragma unroll
        for (int n = 0; n < 4; ++n) acc[m][n] = (v8f){};

    stage(0, 0);
    asm volatile("s_wait_asynccnt 0x0" ::: "memory");
    __syncthreads();

    for (int kt = 0; kt < NKT; ++kt) {
        const int buf = kt & 1;
        if (kt + 1 < NKT) stage((kt + 1) * 32, buf ^ 1);

        // A 16x32 frag: lane(rl,g): elems 0-7 = K 8g..8g+7, elems 8-15 = K 16+8g..
        AFrag a0, a1;
        {
            const bf16_t* ap = &lA[buf][(wRow + rl) * LDT + (g << 3)];
            a0.h[0] = *(const v8bf*)ap; a0.h[1] = *(const v8bf*)(ap + 16);
            ap += 16 * LDT;
            a1.h[0] = *(const v8bf*)ap; a1.h[1] = *(const v8bf*)(ap + 16);
        }
        // B 32x16 frags: lane(rl,g): elems 0-7 = K 8g..8g+7, elems 8-15 = K 16+8g.. for column rl
        #pragma unroll
        for (int n = 0; n < 4; ++n) {
            AFrag bfr;
            const bf16_t* bp = &lB[buf][(wCol + 16 * n + rl) * LDT + (g << 3)];
            bfr.h[0] = *(const v8bf*)bp; bfr.h[1] = *(const v8bf*)(bp + 16);
            acc[0][n] = __builtin_amdgcn_wmma_f32_16x16x32_bf16(false, a0.v, false, bfr.v,
                                                                (short)0, acc[0][n], false, false);
            acc[1][n] = __builtin_amdgcn_wmma_f32_16x16x32_bf16(false, a1.v, false, bfr.v,
                                                                (short)0, acc[1][n], false, false);
        }

        if (kt + 1 < NKT) {
            asm volatile("s_wait_asynccnt 0x0" ::: "memory");
            __syncthreads();
        }
    }

    // ---- epilogue: add bias, store Z (C layout: VGPR e -> row 8g+e, col = rl) ----
    #pragma unroll
    for (int m = 0; m < 2; ++m) {
        #pragma unroll
        for (int n = 0; n < 4; ++n) {
            int col = colBase + wCol + 16 * n + rl;
            float bv = bias[col];
            #pragma unroll
            for (int e = 0; e < 8; ++e) {
                int row = rowBase + wRow + 16 * m + (g << 3) + e;
                Zd[(size_t)row * FH + col] = acc[m][n][e] + bv;
            }
        }
    }
}

// ---------------- gate nonlinearities + state update ----------------
__global__ __launch_bounds__(256)
void lstm_gates(const float* __restrict__ Z, float* __restrict__ Cst,
                bf16_t* __restrict__ Hbf, float* __restrict__ Hcomb)
{
    size_t idx = (size_t)blockIdx.x * blockDim.x + threadIdx.x;   // over 2*BAT*HID
    if (idx >= (size_t)2 * BAT * HID) return;
    int dir = (int)(idx >> 18);          // BAT*HID = 2^18
    int rem = (int)(idx & 262143u);
    int b   = rem >> 10;
    int j   = rem & 1023;
    const float* z = Z + (size_t)dir * BAT * FH + (size_t)b * FH;
    float gg = tanhf(z[j]);
    float ii = 1.f / (1.f + expf(-z[HID + j]));
    float ff = 1.f / (1.f + expf(-z[2 * HID + j]));
    float oo = 1.f / (1.f + expf(-z[3 * HID + j]));
    float cn = gg * ii + Cst[idx] * ff;
    Cst[idx] = cn;
    float h = tanhf(cn) * oo;
    Hbf[idx] = f2bf(h);
    Hcomb[(size_t)b * (2 * HID) + (size_t)dir * HID + j] = h;   // last step's write survives
}

// ---------------- classifier + log_softmax ----------------
__global__ __launch_bounds__(128)
void classifier(const float* __restrict__ Hcomb, const float* __restrict__ pw,
                const float* __restrict__ pb, float* __restrict__ out)
{
    __shared__ float red[NC][128];
    __shared__ float logits[NC];
    const int b = blockIdx.x;
    const int tid = threadIdx.x;
    float acc[NC];
    #pragma unroll
    for (int c = 0; c < NC; ++c) acc[c] = 0.f;
    const float* hrow = Hcomb + (size_t)b * (2 * HID);
    for (int k = tid; k < 2 * HID; k += 128) {
        float hv = hrow[k];
        #pragma unroll
        for (int c = 0; c < NC; ++c) acc[c] += hv * pw[(size_t)c * (2 * HID) + k];
    }
    #pragma unroll
    for (int c = 0; c < NC; ++c) red[c][tid] = acc[c];
    __syncthreads();
    if (tid < NC) {
        float s = 0.f;
        for (int i = 0; i < 128; ++i) s += red[tid][i];
        logits[tid] = s + pb[tid];
    }
    __syncthreads();
    if (tid == 0) {
        float mx = logits[0];
        for (int c = 1; c < NC; ++c) mx = fmaxf(mx, logits[c]);
        float se = 0.f;
        for (int c = 0; c < NC; ++c) se += expf(logits[c] - mx);
        float lse = mx + logf(se);
        for (int c = 0; c < NC; ++c) out[(size_t)b * NC + c] = logits[c] - lse;
    }
}

// ---------------- launch ----------------
extern "C" void kernel_launch(void* const* d_in, const int* in_sizes, int n_in,
                              void* d_out, int out_size, void* d_ws, size_t ws_size,
                              hipStream_t stream) {
    const int*   x     = (const int*)d_in[0];
    const float* embed = (const float*)d_in[1];
    const float* fwd_W = (const float*)d_in[2];
    const float* fwd_b = (const float*)d_in[3];
    const float* bwd_W = (const float*)d_in[4];
    const float* bwd_b = (const float*)d_in[5];
    const float* p_w   = (const float*)d_in[6];
    const float* p_b   = (const float*)d_in[7];
    float* out = (float*)d_out;

    char* ws = (char*)d_ws;
    bf16_t* Wbf   = (bf16_t*)(ws + OFF_W);
    bf16_t* Xbf   = (bf16_t*)(ws + OFF_X);
    bf16_t* Hbf   = (bf16_t*)(ws + OFF_H);
    float*  Cst   = (float*) (ws + OFF_C);
    float*  Z     = (float*) (ws + OFF_Z);
    float*  Hcomb = (float*) (ws + OFF_HC);

    {
        size_t n = (size_t)2 * FH * KK / 8;
        cvt_weights<<<(unsigned)((n + 255) / 256), 256, 0, stream>>>(fwd_W, bwd_W, Wbf);
    }
    {
        size_t n = (size_t)SEQ * BAT * IN / 8;
        gather_embed<<<(unsigned)((n + 255) / 256), 256, 0, stream>>>(x, embed, Xbf);
    }
    {
        size_t n = (size_t)2 * BAT * HID;
        init_state<<<(unsigned)((n + 255) / 256), 256, 0, stream>>>(Cst, Hbf);
    }

    for (int t = 0; t < SEQ; ++t) {
        lstm_step_gemm<<<dim3(FH / 128, BAT / 64, 2), 128, 0, stream>>>(
            Xbf, Hbf, Wbf, fwd_b, bwd_b, Z, t);
        size_t n = (size_t)2 * BAT * HID;
        lstm_gates<<<(unsigned)((n + 255) / 256), 256, 0, stream>>>(Z, Cst, Hbf, Hcomb);
    }

    classifier<<<BAT, 128, 0, stream>>>(Hcomb, p_w, p_b, out);
    (void)in_sizes; (void)n_in; (void)out_size; (void)ws_size;
}
